// CombinedGAT_83459804495956
// MI455X (gfx1250) — matrix operand order repaired and checked by reading, not compiled, not run on hardware
//
#include <hip/hip_runtime.h>
#include <hip/hip_bf16.h>
#include <math.h>

typedef __attribute__((ext_vector_type(16))) __bf16 v16bf;
typedef __attribute__((ext_vector_type(8)))  float  v8f;

#define NNODE 50000
#define NEDGE 800000
#define EP    (NNODE + NEDGE)   /* 850000 edges incl. self loops */
#define HIGH  128
#define LOW   32
#define EMBD  64
#define IN1   192
#define HID   32
#define HEADS 8
#define C1    256               /* HEADS*HID */
#define OUTC  8
#define NEG_SLOPE 0.2f

static __device__ __forceinline__ float lrelu(float x) {
  return x > 0.f ? x : NEG_SLOPE * x;
}
static __device__ __forceinline__ float eluf(float x) {
  return x > 0.f ? x : expm1f(x);
}
// Float atomic max via ordered integer encoding (lowers to native
// global_atomic_max_i32 / global_atomic_min_u32 on gfx1250).
static __device__ __forceinline__ void atomicMaxF(float* addr, float v) {
  if (v >= 0.f) atomicMax((int*)addr, __float_as_int(v));
  else          atomicMin((unsigned int*)addr, __float_as_uint(v));
}

// ---------------- W1 pre-pack into WMMA B-operand layout (bf16) -------------
// B tile (32x16, K x N) per wave: lanes 0-15 hold K={0..7,16..23},
// lanes 16-31 hold K={8..15,24..31}; column N = lane&15.  Tiles: kt in [0,6), nt in [0,16).
__global__ void k_pack_w1(const float* __restrict__ W1, __bf16* __restrict__ w1p) {
  int t = blockIdx.x * blockDim.x + threadIdx.x;
  if (t >= 6 * 16 * 512) return;
  int j    = t & 15;
  int l    = (t >> 4) & 31;
  int tile = t >> 9;              // kt*16 + nt
  int nt   = tile & 15;
  int kt   = tile >> 4;
  int k = kt * 32 + (l >> 4) * 8 + (j & 7) + (j >> 3) * 16;
  int n = nt * 16 + (l & 15);
  w1p[t] = (__bf16)W1[k * C1 + n];
}

// ---------------- Embedding + ELU + concat, cast to bf16 --------------------
__global__ void k_embed(const float* __restrict__ high, const float* __restrict__ low,
                        const float* __restrict__ Wemb, const float* __restrict__ bemb,
                        __bf16* __restrict__ xbf) {
  int t = blockIdx.x * blockDim.x + threadIdx.x;
  if (t >= NNODE * IN1) return;
  int n = t / IN1, j = t - n * IN1;
  float v;
  if (j < HIGH) {
    v = high[n * HIGH + j];
  } else {
    int c = j - HIGH;
    float acc = bemb[c];
    #pragma unroll
    for (int k = 0; k < LOW; ++k) acc += low[n * LOW + k] * Wemb[k * EMBD + c];
    v = eluf(acc);
  }
  xbf[t] = (__bf16)v;
}

// ---------------- Accumulator init ------------------------------------------
__global__ void k_init(float* __restrict__ out1, float* __restrict__ max1,
                       float* __restrict__ sum1, float* __restrict__ out2,
                       float* __restrict__ max2, float* __restrict__ sum2) {
  int t = blockIdx.x * blockDim.x + threadIdx.x;
  if (t < NNODE * C1) out1[t] = 0.f;
  if (t < NNODE * HEADS) { max1[t] = -__builtin_inff(); sum1[t] = 0.f; out2[t] = 0.f; }
  if (t < NNODE) { max2[t] = -__builtin_inff(); sum2[t] = 0.f; }
}

// ---------------- GEMM1: h1[N,256] = x(bf16)[N,192] @ W1(bf16) --------------
// One wave per 16x16 output tile; K loop of 6 x (K=32) WMMA steps.
__global__ void __launch_bounds__(256) k_gemm1(const __bf16* __restrict__ xbf,
                                               const __bf16* __restrict__ w1p,
                                               float* __restrict__ h1) {
  int lane  = threadIdx.x & 31;
  int wid   = blockIdx.x * (blockDim.x >> 5) + (threadIdx.x >> 5);
  int tileM = wid >> 4;          // 3125 row tiles (50000 = 3125*16 exactly)
  int tileN = wid & 15;          // 16 col tiles
  int hi    = lane >> 4;
  int rowA  = tileM * 16 + (lane & 15);
  v8f acc = {};
  #pragma unroll
  for (int kt = 0; kt < 6; ++kt) {
    // A: 16x32 bf16 layout — lane<16: K={0..7,16..23}; lane>=16: K={8..15,24..31}
    const __bf16* pa = xbf + rowA * IN1 + kt * 32 + hi * 8;
    v16bf a, b;
    #pragma unroll
    for (int j = 0; j < 8; ++j) { a[j] = pa[j]; a[j + 8] = pa[16 + j]; }
    const __bf16* pb = w1p + ((kt * 16 + tileN) * 512 + lane * 16);
    #pragma unroll
    for (int j = 0; j < 16; ++j) b[j] = pb[j];
    acc = __builtin_amdgcn_wmma_f32_16x16x32_bf16(false, a, false, b, (short)0, acc,
                                                  false, false);
  }
  // C/D layout: VGPR v -> M = v + 8*hi, N = lane&15
  int col  = tileN * 16 + (lane & 15);
  int row0 = tileM * 16 + hi * 8;
  #pragma unroll
  for (int v = 0; v < 8; ++v) h1[(row0 + v) * C1 + col] = acc[v];
}

// ---------------- Per-node attention logits, layer 1 ------------------------
__global__ void k_att1(const float* __restrict__ h1, const float* __restrict__ as1,
                       const float* __restrict__ ad1, float* __restrict__ a_src,
                       float* __restrict__ a_dst) {
  int t = blockIdx.x * blockDim.x + threadIdx.x;
  if (t >= NNODE * HEADS) return;
  int n = t >> 3, hd = t & 7;
  const float* hp = h1 + n * C1 + hd * HID;
  float s = 0.f, d = 0.f;
  #pragma unroll
  for (int c = 0; c < HID; ++c) {
    float hv = hp[c];
    s += hv * as1[hd * HID + c];
    d += hv * ad1[hd * HID + c];
  }
  a_src[t] = s; a_dst[t] = d;
}

// ---------------- Edge pass A1: segment max ---------------------------------
__global__ void k_edge_max1(const int* __restrict__ ei, const float* __restrict__ a_src,
                            const float* __restrict__ a_dst, float* __restrict__ max1) {
  int t = blockIdx.x * blockDim.x + threadIdx.x;
  if (t >= EP * HEADS) return;
  int e = t >> 3, hd = t & 7;
  int s, d;
  if (e < NEDGE) { s = ei[e]; d = ei[NEDGE + e]; } else { s = d = e - NEDGE; }
  float a = lrelu(a_src[s * 8 + hd] + a_dst[d * 8 + hd]);
  atomicMaxF(&max1[d * 8 + hd], a);
}

// ---------------- Edge pass B1: exp-sum + weighted message scatter ----------
// One wave per edge; lane handles 8 channels (head = lane>>2).
__global__ void __launch_bounds__(256) k_edge_acc1(
    const int* __restrict__ ei, const float* __restrict__ a_src,
    const float* __restrict__ a_dst, const float* __restrict__ max1,
    const float* __restrict__ h1, float* __restrict__ sum1,
    float* __restrict__ out1) {
  int lane = threadIdx.x & 31;
  int e = blockIdx.x * (blockDim.x >> 5) + (threadIdx.x >> 5);
  if (e >= EP) return;
  int s, d;
  if (e < NEDGE) { s = ei[e]; d = ei[NEDGE + e]; } else { s = d = e - NEDGE; }
  int hd = lane >> 2;
  float a  = lrelu(a_src[s * 8 + hd] + a_dst[d * 8 + hd]);
  float ew = expf(a - max1[d * 8 + hd]);
  if ((lane & 3) == 0) atomicAdd(&sum1[d * 8 + hd], ew);
  int ch = lane * 8;
  const float4* hp = (const float4*)(h1 + s * C1 + ch);
  float4 h0 = hp[0], h4 = hp[1];
  float* op = out1 + d * C1 + ch;
  atomicAdd(op + 0, ew * h0.x); atomicAdd(op + 1, ew * h0.y);
  atomicAdd(op + 2, ew * h0.z); atomicAdd(op + 3, ew * h0.w);
  atomicAdd(op + 4, ew * h4.x); atomicAdd(op + 5, ew * h4.y);
  atomicAdd(op + 6, ew * h4.z); atomicAdd(op + 7, ew * h4.w);
}

// ---------------- Finalize layer 1: normalize + bias + ELU -> x2 ------------
__global__ void k_final1(const float* __restrict__ out1, const float* __restrict__ sum1,
                         const float* __restrict__ b1, float* __restrict__ x2) {
  int t = blockIdx.x * blockDim.x + threadIdx.x;
  if (t >= NNODE * C1) return;
  int n = t >> 8, ch = t & 255;
  float v = out1[t] / (sum1[n * 8 + (ch >> 5)] + 1e-16f) + b1[ch];
  x2[t] = eluf(v);
}

// ---------------- GEMM2: h2[N,8] = x2[N,256] @ W2[256,8] (VALU, f32) --------
__global__ void k_h2(const float* __restrict__ x2, const float* __restrict__ W2,
                     float* __restrict__ h2) {
  int t = blockIdx.x * blockDim.x + threadIdx.x;
  if (t >= NNODE * OUTC) return;
  int n = t >> 3, o = t & 7;
  const float* xp = x2 + n * C1;
  float acc = 0.f;
  #pragma unroll 8
  for (int k = 0; k < C1; ++k) acc += xp[k] * W2[k * OUTC + o];
  h2[t] = acc;
}

// ---------------- Per-node attention logits, layer 2 ------------------------
__global__ void k_att2(const float* __restrict__ h2, const float* __restrict__ as2,
                       const float* __restrict__ ad2, float* __restrict__ a_src2,
                       float* __restrict__ a_dst2) {
  int n = blockIdx.x * blockDim.x + threadIdx.x;
  if (n >= NNODE) return;
  float s = 0.f, d = 0.f;
  #pragma unroll
  for (int o = 0; o < OUTC; ++o) {
    float hv = h2[n * OUTC + o];
    s += hv * as2[o];
    d += hv * ad2[o];
  }
  a_src2[n] = s; a_dst2[n] = d;
}

// ---------------- Edge passes, layer 2 (thread per edge) --------------------
__global__ void k_edge_max2(const int* __restrict__ ei, const float* __restrict__ a_src2,
                            const float* __restrict__ a_dst2, float* __restrict__ max2) {
  int e = blockIdx.x * blockDim.x + threadIdx.x;
  if (e >= EP) return;
  int s, d;
  if (e < NEDGE) { s = ei[e]; d = ei[NEDGE + e]; } else { s = d = e - NEDGE; }
  atomicMaxF(&max2[d], lrelu(a_src2[s] + a_dst2[d]));
}

__global__ void k_edge_acc2(const int* __restrict__ ei, const float* __restrict__ a_src2,
                            const float* __restrict__ a_dst2, const float* __restrict__ max2,
                            const float* __restrict__ h2, float* __restrict__ sum2,
                            float* __restrict__ out2) {
  int e = blockIdx.x * blockDim.x + threadIdx.x;
  if (e >= EP) return;
  int s, d;
  if (e < NEDGE) { s = ei[e]; d = ei[NEDGE + e]; } else { s = d = e - NEDGE; }
  float ew = expf(lrelu(a_src2[s] + a_dst2[d]) - max2[d]);
  atomicAdd(&sum2[d], ew);
  const float4* hp = (const float4*)(h2 + s * OUTC);
  float4 h0 = hp[0], h4 = hp[1];
  float* op = out2 + d * OUTC;
  atomicAdd(op + 0, ew * h0.x); atomicAdd(op + 1, ew * h0.y);
  atomicAdd(op + 2, ew * h0.z); atomicAdd(op + 3, ew * h0.w);
  atomicAdd(op + 4, ew * h4.x); atomicAdd(op + 5, ew * h4.y);
  atomicAdd(op + 6, ew * h4.z); atomicAdd(op + 7, ew * h4.w);
}

// ---------------- Final: normalize + bias + log_softmax ---------------------
__global__ void k_out(const float* __restrict__ out2, const float* __restrict__ sum2,
                      const float* __restrict__ b2, float* __restrict__ y) {
  int n = blockIdx.x * blockDim.x + threadIdx.x;
  if (n >= NNODE) return;
  float inv = 1.f / (sum2[n] + 1e-16f);
  float v[OUTC];
  float m = -__builtin_inff();
  #pragma unroll
  for (int o = 0; o < OUTC; ++o) {
    v[o] = out2[n * OUTC + o] * inv + b2[o];
    m = fmaxf(m, v[o]);
  }
  float z = 0.f;
  #pragma unroll
  for (int o = 0; o < OUTC; ++o) z += expf(v[o] - m);
  float l = m + logf(z);
  #pragma unroll
  for (int o = 0; o < OUTC; ++o) y[n * OUTC + o] = v[o] - l;
}

extern "C" void kernel_launch(void* const* d_in, const int* in_sizes, int n_in,
                              void* d_out, int out_size, void* d_ws, size_t ws_size,
                              hipStream_t stream) {
  const float* high = (const float*)d_in[0];
  const float* low  = (const float*)d_in[1];
  const int*   ei   = (const int*)d_in[2];
  const float* Wemb = (const float*)d_in[3];
  const float* bemb = (const float*)d_in[4];
  const float* W1   = (const float*)d_in[5];
  const float* as1  = (const float*)d_in[6];
  const float* ad1  = (const float*)d_in[7];
  const float* b1   = (const float*)d_in[8];
  const float* W2   = (const float*)d_in[9];
  const float* as2  = (const float*)d_in[10];
  const float* ad2  = (const float*)d_in[11];
  const float* b2   = (const float*)d_in[12];
  float* y = (float*)d_out;

  char* base = (char*)d_ws;
  size_t off = 0;
  auto wsalloc = [&](size_t bytes) -> void* {
    void* r = base + off;
    off = (off + bytes + 255) & ~(size_t)255;
    return r;
  };
  __bf16* xbf   = (__bf16*)wsalloc((size_t)NNODE * IN1 * 2);
  __bf16* w1p   = (__bf16*)wsalloc((size_t)6 * 16 * 512 * 2);
  float*  h1    = (float*)wsalloc((size_t)NNODE * C1 * 4);   // later aliased as x2
  float*  out1  = (float*)wsalloc((size_t)NNODE * C1 * 4);
  float*  asrc1 = (float*)wsalloc((size_t)NNODE * HEADS * 4);
  float*  adst1 = (float*)wsalloc((size_t)NNODE * HEADS * 4);
  float*  max1  = (float*)wsalloc((size_t)NNODE * HEADS * 4);
  float*  sum1  = (float*)wsalloc((size_t)NNODE * HEADS * 4);
  float*  h2    = (float*)wsalloc((size_t)NNODE * OUTC * 4);
  float*  out2  = (float*)wsalloc((size_t)NNODE * OUTC * 4);
  float*  asrc2 = (float*)wsalloc((size_t)NNODE * 4);
  float*  adst2 = (float*)wsalloc((size_t)NNODE * 4);
  float*  max2  = (float*)wsalloc((size_t)NNODE * 4);
  float*  sum2  = (float*)wsalloc((size_t)NNODE * 4);
  float*  x2    = h1;   // alias: h1 dead after k_edge_acc1

  const int B = 256;
  auto grid = [](long n) { return (int)((n + 255) / 256); };

  k_pack_w1<<<grid(6 * 16 * 512), B, 0, stream>>>(W1, w1p);
  k_embed<<<grid((long)NNODE * IN1), B, 0, stream>>>(high, low, Wemb, bemb, xbf);
  k_init<<<grid((long)NNODE * C1), B, 0, stream>>>(out1, max1, sum1, out2, max2, sum2);
  k_gemm1<<<(NNODE * 16) / 8, B, 0, stream>>>(xbf, w1p, h1);   // 100000 blocks, 8 waves each
  k_att1<<<grid((long)NNODE * HEADS), B, 0, stream>>>(h1, as1, ad1, asrc1, adst1);
  k_edge_max1<<<grid((long)EP * HEADS), B, 0, stream>>>(ei, asrc1, adst1, max1);
  k_edge_acc1<<<EP / 8, B, 0, stream>>>(ei, asrc1, adst1, max1, h1, sum1, out1);
  k_final1<<<grid((long)NNODE * C1), B, 0, stream>>>(out1, sum1, b1, x2);
  k_h2<<<grid((long)NNODE * OUTC), B, 0, stream>>>(x2, W2, h2);
  k_att2<<<grid(NNODE), B, 0, stream>>>(h2, as2, ad2, asrc2, adst2);
  k_edge_max2<<<grid(EP), B, 0, stream>>>(ei, asrc2, adst2, max2);
  k_edge_acc2<<<grid(EP), B, 0, stream>>>(ei, asrc2, adst2, max2, h2, sum2, out2);
  k_out<<<grid(NNODE), B, 0, stream>>>(out2, sum2, b2, y);
}
